// RBFInterpolator_22325240004942
// MI455X (gfx1250) — compile-verified, matrix operand order checked by reading, not run on hardware
//
#include <hip/hip_runtime.h>

// ---- CDNA5 WMMA types -------------------------------------------------------
typedef __attribute__((ext_vector_type(16))) _Float16 v16h;
typedef __attribute__((ext_vector_type(8)))  float    v8f;

#define NDIM     3
#define NCTR     4096
#define SOUT     64
#define MT       128              // query rows per workgroup
#define KT       64               // centers per pipeline stage (2 wmma K-steps)
#define NITER    (NCTR / KT)      // 64
#define KV_PITCH 68               // halves: 8B-aligned rows, conflict-free frag reads
#define B_PITCH  68

union Frag { v16h v; unsigned int u[8]; };

static __device__ __forceinline__ int kmap(int v) {
    // 16-bit A/B fragment K mapping within one 16-K half (ISA 7.12.2)
    return (v < 4) ? 2 * v : 16 + 2 * (v - 4);
}

static __device__ __forceinline__ unsigned int pk2h(float a, float b) {
    // v_cvt_pk_rtz_f16_f32: two f32 -> packed f16 pair in one VALU op
    auto p = __builtin_amdgcn_cvt_pkrtz(a, b);
    return __builtin_bit_cast(unsigned int, p);
}

static __device__ __forceinline__ float tps(float r2) {
    // 0.5*r2*ln(r2) = (ln2/2)*r2*log2(r2); clamp keeps log input normal so
    // r2==0 gives 0 * finite = 0 exactly (matches reference f(0)=0).
    return 0.34657359f * r2 * __log2f(fmaxf(r2, 1e-37f));
}

__global__ __launch_bounds__(256, 2)
void rbf_tps_wmma(const float* __restrict__ x,
                  const float* __restrict__ y,
                  const float* __restrict__ coeffs,
                  const float* __restrict__ shift,
                  const float* __restrict__ scale,
                  const int*   __restrict__ powers,
                  float* __restrict__ out,
                  int nmono)
{
    __shared__ float xT[MT][4];
    __shared__ __align__(16) unsigned short kvT[2][MT * KV_PITCH];
    __shared__ __align__(16) unsigned short bT[2][SOUT * B_PITCH];

    const int t    = threadIdx.x;
    const int lane = t & 31;           // wave32
    const int wave = t >> 5;           // 0..7
    const int mBlk = blockIdx.x * MT;

    // ---- stage query tile ---------------------------------------------------
    if (t < MT) {
        const float* xp = &x[(mBlk + t) * 3];
        xT[t][0] = xp[0];
        xT[t][1] = xp[1];
        xT[t][2] = xp[2];
    }
    __syncthreads();

    // ---- hoist this thread's 8 query rows into registers for the K-loop -----
    const int mBase = t >> 4;          // 0..15 ; rows mBase + 16*i
    const int jj4   = (t & 15) * 4;    // 4 consecutive centers per thread
    float xr0[8], xr1[8], xr2[8];
    #pragma unroll
    for (int i = 0; i < 8; ++i) {
        int m = mBase + 16 * i;
        xr0[i] = xT[m][0];
        xr1[i] = xT[m][1];
        xr2[i] = xT[m][2];
    }

    // Stage one K-step (B tile f32->f16 N-major; 128x64 TPS tile) into slot.
    auto stage = [&](unsigned short* kvB, unsigned short* bB, int k0) {
        // ---- B tile: pack two k-adjacent coeffs per b32 store ----
        #pragma unroll
        for (int e = 0; e < 8; ++e) {
            int g   = t + 256 * e;         // 0..2047
            int kk2 = (g >> 6) * 2;        // 0,2,..,62
            int nn  = g & 63;
            float c0 = coeffs[(k0 + kk2)     * SOUT + nn];
            float c1 = coeffs[(k0 + kk2 + 1) * SOUT + nn];
            *reinterpret_cast<unsigned int*>(&bB[nn * B_PITCH + kk2]) = pk2h(c0, c1);
        }
        // ---- kv tile: 8 rows x 4 centers per thread ----
        const float* yp = &y[(size_t)(k0 + jj4) * 3];  // 12 floats, 16B aligned
        float4 ya = *reinterpret_cast<const float4*>(yp + 0);
        float4 yb = *reinterpret_cast<const float4*>(yp + 4);
        float4 yc = *reinterpret_cast<const float4*>(yp + 8);
        float c0x = ya.x, c0y = ya.y, c0z = ya.z;
        float c1x = ya.w, c1y = yb.x, c1z = yb.y;
        float c2x = yb.z, c2y = yb.w, c2z = yc.x;
        float c3x = yc.y, c3y = yc.z, c3z = yc.w;
        #pragma unroll
        for (int i = 0; i < 8; ++i) {
            int m = mBase + 16 * i;
            float x0 = xr0[i], x1 = xr1[i], x2 = xr2[i];
            float d0, d1, d2;
            d0 = x0 - c0x; d1 = x1 - c0y; d2 = x2 - c0z;
            float r20 = d0 * d0 + d1 * d1 + d2 * d2;
            d0 = x0 - c1x; d1 = x1 - c1y; d2 = x2 - c1z;
            float r21 = d0 * d0 + d1 * d1 + d2 * d2;
            d0 = x0 - c2x; d1 = x1 - c2y; d2 = x2 - c2z;
            float r22 = d0 * d0 + d1 * d1 + d2 * d2;
            d0 = x0 - c3x; d1 = x1 - c3y; d2 = x2 - c3z;
            float r23 = d0 * d0 + d1 * d1 + d2 * d2;
            uint2 pk;
            pk.x = pk2h(tps(r20), tps(r21));
            pk.y = pk2h(tps(r22), tps(r23));
            *reinterpret_cast<uint2*>(&kvB[m * KV_PITCH + jj4]) = pk;  // ds_store_b64
        }
    };

    // ---- prologue: fill pipeline stage 0 ------------------------------------
    stage(kvT[0], bT[0], 0);
    __syncthreads();

    v8f acc0 = {}, acc1 = {}, acc2 = {}, acc3 = {};

    const int nl   = lane & 15;
    const int kb16 = (lane >> 4) << 3;          // K offset 0 / 8 within 16-K half
    const int row  = wave * 16 + nl;

    for (int it = 0; it < NITER; ++it) {
        const int cur = it & 1;
        const unsigned short* kvC = kvT[cur];
        const unsigned short* bC  = bT[cur];

        // prefetch coeff tile two stages ahead into L2
        if (it + 2 < NITER)
            __builtin_prefetch(&coeffs[(it + 2) * KT * SOUT + (t & 63)], 0, 0);

        // ---- consume current tile: 8 WMMAs (issued first so the staging -----
        // ---- VALU/log work below co-executes while the matrix pipe drains) --
        #pragma unroll
        for (int ks = 0; ks < KT; ks += 32) {
            Frag a;
            #pragma unroll
            for (int v = 0; v < 8; ++v) {
                a.u[v] = *reinterpret_cast<const unsigned int*>(
                             &kvC[row * KV_PITCH + kmap(v) + kb16 + ks]);
            }
            #pragma unroll
            for (int nf = 0; nf < 4; ++nf) {
                int n = nf * 16 + nl;
                Frag b;
                #pragma unroll
                for (int v = 0; v < 8; ++v) {
                    b.u[v] = *reinterpret_cast<const unsigned int*>(
                                 &bC[n * B_PITCH + kmap(v) + kb16 + ks]);
                }
                if (nf == 0)
                    acc0 = __builtin_amdgcn_wmma_f32_16x16x32_f16(
                        false, a.v, false, b.v, (short)0, acc0, false, false);
                else if (nf == 1)
                    acc1 = __builtin_amdgcn_wmma_f32_16x16x32_f16(
                        false, a.v, false, b.v, (short)0, acc1, false, false);
                else if (nf == 2)
                    acc2 = __builtin_amdgcn_wmma_f32_16x16x32_f16(
                        false, a.v, false, b.v, (short)0, acc2, false, false);
                else
                    acc3 = __builtin_amdgcn_wmma_f32_16x16x32_f16(
                        false, a.v, false, b.v, (short)0, acc3, false, false);
            }
        }

        // ---- produce next tile into the other buffer ------------------------
        if (it + 1 < NITER)
            stage(kvT[cur ^ 1], bT[cur ^ 1], (it + 1) * KT);

        __syncthreads();
    }

    // ---- epilogue: polynomial tail + store ----------------------------------
    {
        float sh0 = shift[0], sh1 = shift[1], sh2 = shift[2];
        float sc0 = scale[0], sc1 = scale[1], sc2 = scale[2];
        const int rOff = (lane >> 4) * 8;       // C/D layout: lanes 16-31 -> M+8
        const int m0   = wave * 16;
        const float* ctail = coeffs + (size_t)NCTR * SOUT;

        if (nmono == 4) {
            // degree-1, ndim=3: monomials are [1, x0, x1, x2]
            float c00 = ctail[0 * SOUT +  0 + nl], c01 = ctail[1 * SOUT +  0 + nl],
                  c02 = ctail[2 * SOUT +  0 + nl], c03 = ctail[3 * SOUT +  0 + nl];
            float c10 = ctail[0 * SOUT + 16 + nl], c11 = ctail[1 * SOUT + 16 + nl],
                  c12 = ctail[2 * SOUT + 16 + nl], c13 = ctail[3 * SOUT + 16 + nl];
            float c20 = ctail[0 * SOUT + 32 + nl], c21 = ctail[1 * SOUT + 32 + nl],
                  c22 = ctail[2 * SOUT + 32 + nl], c23 = ctail[3 * SOUT + 32 + nl];
            float c30 = ctail[0 * SOUT + 48 + nl], c31 = ctail[1 * SOUT + 48 + nl],
                  c32 = ctail[2 * SOUT + 48 + nl], c33 = ctail[3 * SOUT + 48 + nl];
            #pragma unroll
            for (int r = 0; r < 8; ++r) {
                int m = m0 + r + rOff;
                float xh0 = (xT[m][0] - sh0) / sc0;
                float xh1 = (xT[m][1] - sh1) / sc1;
                float xh2 = (xT[m][2] - sh2) / sc2;
                float* op = &out[(size_t)(mBlk + m) * SOUT];
                op[ 0 + nl] = acc0[r] + c00 + xh0 * c01 + xh1 * c02 + xh2 * c03;
                op[16 + nl] = acc1[r] + c10 + xh0 * c11 + xh1 * c12 + xh2 * c13;
                op[32 + nl] = acc2[r] + c20 + xh0 * c21 + xh1 * c22 + xh2 * c23;
                op[48 + nl] = acc3[r] + c30 + xh0 * c31 + xh1 * c32 + xh2 * c33;
            }
        } else {
            // generic powers-driven path
            #pragma unroll
            for (int r = 0; r < 8; ++r) {
                int m = m0 + r + rOff;
                float xh[3];
                xh[0] = (xT[m][0] - sh0) / sc0;
                xh[1] = (xT[m][1] - sh1) / sc1;
                xh[2] = (xT[m][2] - sh2) / sc2;
                float add0 = 0.f, add1 = 0.f, add2 = 0.f, add3 = 0.f;
                for (int j = 0; j < nmono; ++j) {
                    float pm = 1.f;
                    #pragma unroll
                    for (int d = 0; d < NDIM; ++d) {
                        int e = powers[j * NDIM + d];
                        while (e-- > 0) pm *= xh[d];
                    }
                    const float* cr = &ctail[j * SOUT];
                    add0 += pm * cr[ 0 + nl];
                    add1 += pm * cr[16 + nl];
                    add2 += pm * cr[32 + nl];
                    add3 += pm * cr[48 + nl];
                }
                float* op = &out[(size_t)(mBlk + m) * SOUT];
                op[ 0 + nl] = acc0[r] + add0;
                op[16 + nl] = acc1[r] + add1;
                op[32 + nl] = acc2[r] + add2;
                op[48 + nl] = acc3[r] + add3;
            }
        }
    }
}

extern "C" void kernel_launch(void* const* d_in, const int* in_sizes, int n_in,
                              void* d_out, int out_size, void* d_ws, size_t ws_size,
                              hipStream_t stream) {
    const float* x      = (const float*)d_in[0];
    const float* y      = (const float*)d_in[1];
    const float* coeffs = (const float*)d_in[2];
    const float* shift  = (const float*)d_in[3];
    const float* scale  = (const float*)d_in[4];
    const int*   powers = (const int*)d_in[5];
    float* out = (float*)d_out;

    int nq    = in_sizes[0] / NDIM;    // 65536
    int nmono = in_sizes[5] / NDIM;    // 4
    dim3 grid(nq / MT);                 // 512 workgroups
    rbf_tps_wmma<<<grid, 256, 0, stream>>>(x, y, coeffs, shift, scale, powers,
                                           out, nmono);
}
